// SpikingHabenula_16673063043182
// MI455X (gfx1250) — compile-verified, matrix operand order checked by reading, not run on hardware
//
#include <hip/hip_runtime.h>
#include <cstddef>

typedef __attribute__((ext_vector_type(4)))  float    v4f;
typedef __attribute__((ext_vector_type(2)))  float    v2f;
typedef __attribute__((ext_vector_type(8)))  float    v8f;
typedef __attribute__((ext_vector_type(16))) _Float16 v16h;

#define T_STEPS 20
#define P_A     0.02f
#define P_B     0.2f
#define P_C     (-65.0f)
#define P_D     8.0f
#define V_PEAK  30.0f
#define I_TONIC (-1.0f)
#define R_DECAY 0.9f
#define R_GAIN  0.1f   // float32(1.0 - 0.9) == 0.1f

// ---------------------------------------------------------------------------
// Wave32 sum via WMMA: A holds the 32 lane values (16x4 f32 A layout puts lane
// L in row L%16), B = all-ones => D[m][n] = x[m] + x[m+16]. Each lane sums its
// 8 D VGPRs (rows 0-7 or 8-15 of its column), one shfl_xor(16) finishes it.
// ---------------------------------------------------------------------------
__device__ __forceinline__ float wave_sum32(float x) {
#if __has_builtin(__builtin_amdgcn_wmma_f32_16x16x4_f32)
  v2f a; a.x = x;    a.y = 0.0f;
  v2f b; b.x = 1.0f; b.y = 1.0f;
  v8f c = {};
  c = __builtin_amdgcn_wmma_f32_16x16x4_f32(false, a, false, b,
                                            (short)0, c, false, false);
  float s = c[0]+c[1]+c[2]+c[3]+c[4]+c[5]+c[6]+c[7];
  s += __shfl_xor(s, 16);
  return s;
#elif __has_builtin(__builtin_amdgcn_wmma_f32_16x16x32_f16)
  // 16-bit A layout: lane L -> row L%16 (lanes 0-15 at K=0, 16-31 at K=8).
  v16h a = {}; a[0] = (_Float16)x;
  v16h b;
  #pragma unroll
  for (int i = 0; i < 16; ++i) b[i] = (_Float16)1.0f;
  v8f c = {};
  c = __builtin_amdgcn_wmma_f32_16x16x32_f16(false, a, false, b,
                                             (short)0, c, false, false);
  float s = c[0]+c[1]+c[2]+c[3]+c[4]+c[5]+c[6]+c[7];
  s += __shfl_xor(s, 16);
  return s;
#else
  #pragma unroll
  for (int off = 16; off > 0; off >>= 1) x += __shfl_xor(x, off);
  return x;
#endif
}

__device__ __forceinline__ void izh_step(float& v, float& u, float& r,
                                         float n, float idrive) {
  float I  = idrive + I_TONIC + n * 0.5f;
  float v1 = v + (0.04f * v * v + 5.0f * v + 140.0f - u + I);
  float u1 = u + P_A * (P_B * v - u);
  bool  sp = v1 >= V_PEAK;
  v = sp ? P_C : v1;
  u = sp ? (u1 + P_D) : u1;
  r = r * R_DECAY + (sp ? R_GAIN : 0.0f);
}

// ---------------------------------------------------------------------------
// Streaming kernel: each thread owns 4 neurons, carries v/u/rate in registers
// across the fully unrolled T loop, streams noise non-temporally (b128),
// then WMMA wave-reduce + LDS cross-wave reduce -> one partial per block.
// mode 0 (LHB): idrive = max(0, -(s0-s1)) * 20 ; mode 1 (MHB): idrive = s0*15.
// ---------------------------------------------------------------------------
__global__ __launch_bounds__(256) void izh_stream_kernel(
    const v4f* __restrict__ v_in, const v4f* __restrict__ u_in,
    const v4f* __restrict__ r_in, const v4f* __restrict__ noise,
    const float* __restrict__ s0, const float* __restrict__ s1,
    float* __restrict__ partials, int nq, int mode)
{
  const int tid = blockIdx.x * blockDim.x + threadIdx.x;
  const float idrive = (mode == 0)
      ? fmaxf(0.0f, -(s0[0] - s1[0])) * 20.0f
      : s0[0] * 15.0f;

  float partial = 0.0f;
  if (tid < nq) {
    const v4f vv = v_in[tid];
    const v4f uu = u_in[tid];
    const v4f rr = r_in[tid];
    float v[4] = { vv.x, vv.y, vv.z, vv.w };
    float u[4] = { uu.x, uu.y, uu.z, uu.w };
    float r[4] = { rr.x, rr.y, rr.z, rr.w };
    #pragma unroll
    for (int t = 0; t < T_STEPS; ++t) {
      v4f nvec = __builtin_nontemporal_load(&noise[(size_t)t * (size_t)nq + tid]);
      float n[4] = { nvec.x, nvec.y, nvec.z, nvec.w };
      #pragma unroll
      for (int j = 0; j < 4; ++j)
        izh_step(v[j], u[j], r[j], n[j], idrive);
    }
    partial = r[0] + r[1] + r[2] + r[3];
  }

  // All 256 threads reach here (EXEC all-ones for the WMMA).
  __shared__ float wacc[8];
  float ws = wave_sum32(partial);
  const int lane = threadIdx.x & 31;
  const int wv   = threadIdx.x >> 5;
  if (lane == 0) wacc[wv] = ws;
  __syncthreads();
  if (threadIdx.x == 0) {
    float t = 0.0f;
    #pragma unroll
    for (int i = 0; i < 8; ++i) t += wacc[i];
    partials[blockIdx.x] = t;
  }
}

// ---------------------------------------------------------------------------
// Finalize: deterministic fixed-order reduction of block partials + the
// scalar frustration/helplessness bookkeeping. Single block.
// ---------------------------------------------------------------------------
__global__ __launch_bounds__(256) void izh_finalize_kernel(
    const float* __restrict__ partials, int nbL, int nbM,
    const float* __restrict__ reward_p, const float* __restrict__ expected_p,
    const float* __restrict__ aversion_p, const float* __restrict__ da_p,
    const int* __restrict__ goal_p, const float* __restrict__ frustration_p,
    float* __restrict__ out, float invNL, float invNM)
{
  __shared__ float redL[256];
  __shared__ float redM[256];
  float sl = 0.0f, sm = 0.0f;
  for (int i = threadIdx.x; i < nbL; i += 256) sl += partials[i];
  for (int i = threadIdx.x; i < nbM; i += 256) sm += partials[nbL + i];
  redL[threadIdx.x] = sl;
  redM[threadIdx.x] = sm;
  __syncthreads();
  #pragma unroll
  for (int s = 128; s > 0; s >>= 1) {
    if ((int)threadIdx.x < s) {
      redL[threadIdx.x] += redL[threadIdx.x + s];
      redM[threadIdx.x] += redM[threadIdx.x + s];
    }
    __syncthreads();
  }

  if (threadIdx.x == 0) {
    const float lhb_mean = redL[0] * invNL;
    const float mhb_mean = redM[0] * invNM;

    const float rpe        = reward_p[0] - expected_p[0];
    const float disappoint = fmaxf(0.0f, -rpe);
    const float da_supp    = fminf(0.5f, lhb_mean * 5.0f);
    const float ht5_supp   = fminf(0.3f, lhb_mean * 3.0f);

    const int goal = goal_p[0];
    float f[4];
    #pragma unroll
    for (int i = 0; i < 4; ++i) f[i] = frustration_p[i] * 0.998f;
    if (rpe < -0.05f) f[goal] += 0.05f * fabsf(rpe);
    if (rpe >  0.1f)  f[goal] *= 0.8f;
    #pragma unroll
    for (int i = 0; i < 4; ++i) f[i] = fminf(1.0f, fmaxf(0.0f, f[i]));
    const float helplessness = fmaxf(fmaxf(f[0], f[1]), fmaxf(f[2], f[3]));
    const float sw = (f[goal] > 0.4f) ? 1.0f : 0.0f;
    if (sw > 0.0f) f[goal] *= 0.3f;

    const float dopa_mod = fmaxf(0.5f, 2.0f * (1.0f - da_p[0]));
    const float explore  = fminf(1.0f, disappoint * 2.0f + lhb_mean * 3.0f);

    out[0] = disappoint;
    out[1] = da_supp;
    out[2] = ht5_supp;
    out[3] = lhb_mean;
    out[4] = mhb_mean;
    out[5] = explore;
    out[6] = helplessness;
    out[7] = sw;
    #pragma unroll
    for (int i = 0; i < 4; ++i) out[8 + i]  = f[i] * 0.5f * dopa_mod;
    #pragma unroll
    for (int i = 0; i < 4; ++i) out[12 + i] = f[i];
  }
}

extern "C" void kernel_launch(void* const* d_in, const int* in_sizes, int n_in,
                              void* d_out, int out_size, void* d_ws, size_t ws_size,
                              hipStream_t stream) {
  (void)n_in; (void)out_size; (void)ws_size;
  // Inputs (setup_inputs dict order):
  const float* reward   = (const float*)d_in[0];
  const float* expected = (const float*)d_in[1];
  const float* aversion = (const float*)d_in[2];
  const float* da       = (const float*)d_in[3];
  const int*   goal     = (const int*)  d_in[4];
  const float* frus     = (const float*)d_in[5];
  const v4f*   v_lhb    = (const v4f*)  d_in[6];
  const v4f*   u_lhb    = (const v4f*)  d_in[7];
  const v4f*   r_lhb    = (const v4f*)  d_in[8];
  const v4f*   v_mhb    = (const v4f*)  d_in[9];
  const v4f*   u_mhb    = (const v4f*)  d_in[10];
  const v4f*   r_mhb    = (const v4f*)  d_in[11];
  const v4f*   noise_l  = (const v4f*)  d_in[12];
  const v4f*   noise_m  = (const v4f*)  d_in[13];

  const int nL  = in_sizes[6];          // 2097152
  const int nM  = in_sizes[9];          // 1048576
  const int nqL = nL / 4;
  const int nqM = nM / 4;
  const int nbL = (nqL + 255) / 256;    // 2048
  const int nbM = (nqM + 255) / 256;    // 1024

  float* partials = (float*)d_ws;       // nbL + nbM floats (12 KB)

  izh_stream_kernel<<<nbL, 256, 0, stream>>>(
      v_lhb, u_lhb, r_lhb, noise_l, reward, expected, partials, nqL, 0);
  izh_stream_kernel<<<nbM, 256, 0, stream>>>(
      v_mhb, u_mhb, r_mhb, noise_m, aversion, aversion, partials + nbL, nqM, 1);
  izh_finalize_kernel<<<1, 256, 0, stream>>>(
      partials, nbL, nbM, reward, expected, aversion, da, goal, frus,
      (float*)d_out, 1.0f / (float)nL, 1.0f / (float)nM);
}
